// ConvFauxLarsen_9904194584612
// MI455X (gfx1250) — matrix operand order repaired and checked by reading, not compiled, not run on hardware
//
#include <hip/hip_runtime.h>
#include <hip/hip_bf16.h>
#include <math.h>

// ConvFauxLarsen for MI455X (gfx1250): bf16 WMMA implicit-GEMM convs,
// f32 accumulate, batchnorm as stats+apply kernel pairs, 32-step scan.

#define B      8
#define CC     64
#define ZLEN_  194
#define FSTRIDE 8200

typedef __attribute__((ext_vector_type(16))) __bf16 v16bf;
typedef __attribute__((ext_vector_type(8)))  float  v8f;

__device__ __forceinline__ float cfl_gelu(float x) {
    // jax.nn.gelu default (tanh approximation)
    return 0.5f * x * (1.0f + tanhf(0.7978845608028654f * (x + 0.044715f * x * x * x)));
}
__device__ __forceinline__ unsigned short cfl_f2bf(float f) {
    unsigned int u = __float_as_uint(f);
    return (unsigned short)((u + 0x7fffu + ((u >> 16) & 1u)) >> 16);  // RNE
}

// ---------------------------------------------------------------- expand 1->64
__global__ __launch_bounds__(256)
void k_expand(const float* __restrict__ src, int srcStride, int srcOff, int L,
              const float* __restrict__ w_in, float* __restrict__ out)
{
    int i = blockIdx.x * 256 + threadIdx.x;
    int total = B * L * CC;
    if (i >= total) return;
    int c = i & (CC - 1);
    int t = (i / CC) % L;
    int b = i / (CC * L);
    out[i] = src[(size_t)b * srcStride + srcOff + t] * w_in[c];
}

// ------------------------------------------------------- per-channel BN stats
__global__ __launch_bounds__(256)
void k_stats(const float* __restrict__ in, int N, int C, float* __restrict__ stat)
{
    __shared__ float ss[256];
    __shared__ float sq[256];
    int c = blockIdx.x;
    float s = 0.f, q = 0.f;
    for (int i = threadIdx.x; i < N; i += 256) {
        float v = in[(size_t)i * C + c];
        s += v; q += v * v;
    }
    ss[threadIdx.x] = s; sq[threadIdx.x] = q;
    __syncthreads();
    for (int d = 128; d > 0; d >>= 1) {
        if ((int)threadIdx.x < d) {
            ss[threadIdx.x] += ss[threadIdx.x + d];
            sq[threadIdx.x] += sq[threadIdx.x + d];
        }
        __syncthreads();
    }
    if (threadIdx.x == 0) {
        float m = ss[0] / (float)N;
        float v = sq[0] / (float)N - m * m;
        stat[2 * c]     = m;
        stat[2 * c + 1] = rsqrtf(fmaxf(v, 0.f) + 1e-5f);
    }
}

// ----------------------------------------------- BN + gelu -> f32 + bf16 copy
__global__ __launch_bounds__(256)
void k_bn_gelu(const float* __restrict__ in, const float* __restrict__ stat,
               const float* __restrict__ sc, const float* __restrict__ bi,
               float* __restrict__ outf, unsigned short* __restrict__ outb,
               int total, int Cn)
{
    int i = blockIdx.x * 256 + threadIdx.x;
    if (i >= total) return;
    int c = i % Cn;
    float g = cfl_gelu(sc[c] * (in[i] - stat[2 * c]) * stat[2 * c + 1] + bi[c]);
    outf[i] = g;
    outb[i] = cfl_f2bf(g);
}

// ------------------------------------- BN + gelu + weighted skip (y residual)
__global__ __launch_bounds__(256)
void k_bn_gelu_skip(const float* __restrict__ cv, const float* __restrict__ stat,
                    const float* __restrict__ sc, const float* __restrict__ bi,
                    const float* __restrict__ z, int Lz, int Ly,
                    const float* __restrict__ wsk, float* __restrict__ y)
{
    int i = blockIdx.x * 256 + threadIdx.x;
    int total = B * Ly * CC;
    if (i >= total) return;
    int c = i & (CC - 1);
    int t = (i >> 6) % Ly;
    int b = i / (Ly * CC);
    float g = cfl_gelu(sc[c] * (cv[i] - stat[2 * c]) * stat[2 * c + 1] + bi[c]);
    y[i] = g + z[((size_t)b * Lz + (t + Lz - Ly)) * CC + c] * wsk[c];
}

// -------------------------------------------------------------- conv via WMMA
// in:  (B, Lin, 64) bf16, wT: (64, KW*64) bf16 pre-transposed, out: (B, Lout, 64) f32
template<int KW, int STRIDE>
__global__ __launch_bounds__(128)
void k_conv_wmma(const unsigned short* __restrict__ inb,
                 const unsigned short* __restrict__ wT,
                 float* __restrict__ out, int Lin, int Lout)
{
    constexpr int K    = KW * CC;
    constexpr int SPAN = 16 * STRIDE + KW - 1;
    __shared__ __attribute__((aligned(16))) unsigned short sA[SPAN * CC];

    const int b   = blockIdx.y;
    const int t0  = blockIdx.x * 16;
    const int tid = threadIdx.x;
    const int p0  = t0 * STRIDE;

    // Stage the activation span into LDS (zero-fill past Lin), 4B granules.
    const unsigned int* gsrc = (const unsigned int*)(inb + (size_t)b * Lin * CC);
    unsigned int* sdst = (unsigned int*)sA;
    constexpr int NW = SPAN * CC / 2;          // CC/2 == 32 uints per position
    for (int w = tid; w < NW; w += 128) {
        int p = w >> 5;
        unsigned int v = 0u;
        if (p0 + p < Lin) v = gsrc[(size_t)(p0 + p) * 32 + (w & 31)];
        sdst[w] = v;
    }
    __syncthreads();

    const int lane = tid & 31;
    const int wv   = tid >> 5;     // 4 waves -> 4 x 16 out-channel tiles
    const int n0   = wv * 16;
    const int mn   = lane & 15;    // A row / B,D column
    const int h    = lane >> 4;

    v8f acc = (v8f)(0.0f);
    const unsigned short* wrow = wT + (size_t)(n0 + mn) * K;

    for (int ks = 0; ks < K; ks += 32) {
        union { v16bf v; uint4 q[2]; } a, bm;
        // A fragment (16x32 bf16): lane mn holds K=[ks+8h, +8) and [ks+16+8h, +8)
        int q0 = ks + 8 * h;
        int q1 = ks + 16 + 8 * h;
        a.q[0] = *(const uint4*)&sA[(mn * STRIDE + (q0 >> 6)) * CC + (q0 & 63)];
        a.q[1] = *(const uint4*)&sA[(mn * STRIDE + (q1 >> 6)) * CC + (q1 & 63)];
        // B fragment (32x16 bf16): lane holds column n0+mn, K=[ks+8h, +8) and [ks+16+8h, +8)
        const unsigned short* bp = wrow + ks + 8 * h;
        bm.q[0] = *(const uint4*)(bp);
        bm.q[1] = *(const uint4*)(bp + 16);
        if (ks + 64 < K) __builtin_prefetch((const void*)(wrow + ks + 64), 0, 0);
        acc = __builtin_amdgcn_wmma_f32_16x16x32_bf16(
            false, a.v, false, bm.v, (short)0, acc, false, false);
    }

    // D (16x16 f32): lane = n + 16*(m>=8), VGPR v = m&7
    float* orow = out + (size_t)b * Lout * CC;
    #pragma unroll
    for (int v = 0; v < 8; ++v) {
        int t = t0 + 8 * h + v;
        if (t < Lout) orow[(size_t)t * CC + (n0 + mn)] = acc[v];
    }
}

// --------------------------------------------------------- 64->1 output conv
__global__ __launch_bounds__(256)
void k_outconv(const float* __restrict__ z, int total,
               const float* __restrict__ wo, float* __restrict__ o)
{
    int i = blockIdx.x * 256 + threadIdx.x;
    if (i >= total) return;
    const float* p = z + (size_t)i * CC;
    float s = 0.f;
    #pragma unroll
    for (int c = 0; c < CC; ++c) s += p[c] * wo[c];
    o[i] = s;
}

// ------------------------------------------------------------ final scalar BN
__global__ __launch_bounds__(256)
void k_bn_final(const float* __restrict__ o, const float* __restrict__ stat,
                const float* __restrict__ so, const float* __restrict__ bo,
                float* __restrict__ dst, int dstStride, int dstOff, int Lf)
{
    int i = blockIdx.x * 256 + threadIdx.x;
    if (i >= B * Lf) return;
    int b = i / Lf, t = i % Lf;
    dst[(size_t)b * dstStride + dstOff + t] = so[0] * (o[i] - stat[0]) * stat[1] + bo[0];
}

// -------------------------------------------- weight transpose+bf16: [co][K]
__global__ __launch_bounds__(256)
void k_prep_w(const float* __restrict__ w, int kw, unsigned short* __restrict__ wT)
{
    int i = blockIdx.x * 256 + threadIdx.x;
    int total = kw * CC * CC;
    if (i >= total) return;
    int co   = i & 63;        // w is [tap][ci][co]
    int rest = i >> 6;        // tap*64 + ci
    wT[(size_t)co * (kw * CC) + rest] = cfl_f2bf(w[i]);
}

// ------------------------------------------------------------- scan plumbing
__global__ __launch_bounds__(256)
void k_init_fbuf(const float* __restrict__ x, const float* __restrict__ out,
                 float* __restrict__ fbuf)
{   // foundry = concat(x_masked, z0_last)[-8128:] = x[1..8127] ++ z0_last
    int i = blockIdx.x * 256 + threadIdx.x;
    if (i >= B * 8128) return;
    int b = i / 8128, j = i % 8128;
    fbuf[(size_t)b * FSTRIDE + j] =
        (j < 8127) ? x[(size_t)b * 8192 + 1 + j] : out[(size_t)b * 4000 + 3967];
}

__global__ __launch_bounds__(256)
void k_build_window(float* __restrict__ win, const float* __restrict__ fbuf,
                    int cur_len, const float* __restrict__ x, int step)
{   // window = f[cur-193 .. cur-2], ipt, f[cur-1]
    int i = blockIdx.x * 256 + threadIdx.x;
    if (i >= B * ZLEN_) return;
    int b = i / ZLEN_, j = i % ZLEN_;
    const float* f = fbuf + (size_t)b * FSTRIDE;
    float v;
    if (j < 192)       v = f[cur_len - 193 + j];
    else if (j == 192) v = x[(size_t)b * 8192 + 8128 + 2 * step];
    else               v = f[cur_len - 1];
    win[i] = v;
}

__global__
void k_commit(float* __restrict__ fbuf, int cur_len, const float* __restrict__ x,
              int step, const float* __restrict__ zval, float* __restrict__ out)
{   // f <- [..., ipt, old_last, z]; also emit z1[step]
    int b = threadIdx.x;
    if (b >= B) return;
    float* f  = fbuf + (size_t)b * FSTRIDE;
    float ipt = x[(size_t)b * 8192 + 8128 + 2 * step];
    float tmp = f[cur_len - 1];
    f[cur_len - 1] = ipt;
    f[cur_len]     = tmp;
    f[cur_len + 1] = zval[b];
    out[(size_t)b * 4000 + 3968 + step] = zval[b];
}

// ---------------------------------------------------------------- host side
struct Ctx {
    const float *w_in, *s_in, *b_in, *s0a, *b0a, *s0b, *b0b;
    const float *sa, *ba, *sb, *bb, *w_skip, *w_out, *s_out, *b_out;
    const unsigned short *wT0, *wTc;
    float *stat, *conv, *zf, *obuf;
    unsigned short *zb;
};

static inline int ceil_div(int a, int b) { return (a + b - 1) / b; }

static void run_stack(hipStream_t s, const Ctx& C_,
                      const float* src, int srcStride, int srcOff, int L,
                      float* dst, int dstStride, int dstOff)
{
    int total = B * L * CC;
    k_expand<<<ceil_div(total, 256), 256, 0, s>>>(src, srcStride, srcOff, L, C_.w_in, C_.conv);
    k_stats<<<CC, 256, 0, s>>>(C_.conv, B * L, CC, C_.stat);
    k_bn_gelu<<<ceil_div(total, 256), 256, 0, s>>>(C_.conv, C_.stat, C_.s_in, C_.b_in,
                                                   C_.zf, C_.zb, total, CC);
    int L1 = (L - 14) / 2 + 1;
    k_conv_wmma<14, 2><<<dim3(ceil_div(L1, 16), B), 128, 0, s>>>(C_.zb, C_.wT0, C_.conv, L, L1);
    total = B * L1 * CC;
    k_stats<<<CC, 256, 0, s>>>(C_.conv, B * L1, CC, C_.stat);
    k_bn_gelu<<<ceil_div(total, 256), 256, 0, s>>>(C_.conv, C_.stat, C_.s0a, C_.b0a,
                                                   C_.zf, C_.zb, total, CC);
    k_stats<<<CC, 256, 0, s>>>(C_.zf, B * L1, CC, C_.stat);
    k_bn_gelu<<<ceil_div(total, 256), 256, 0, s>>>(C_.zf, C_.stat, C_.s0b, C_.b0b,
                                                   C_.zf, C_.zb, total, CC);
    int Lc = L1;
    for (int i = 0; i < 15; ++i) {
        int Ly = Lc - 6;
        int ty = B * Ly * CC;
        k_conv_wmma<7, 1><<<dim3(ceil_div(Ly, 16), B), 128, 0, s>>>(
            C_.zb, C_.wTc + (size_t)i * CC * 7 * CC, C_.conv, Lc, Ly);
        k_stats<<<CC, 256, 0, s>>>(C_.conv, B * Ly, CC, C_.stat);
        k_bn_gelu_skip<<<ceil_div(ty, 256), 256, 0, s>>>(
            C_.conv, C_.stat, C_.sa + i * CC, C_.ba + i * CC,
            C_.zf, Lc, Ly, C_.w_skip + i * CC, C_.conv);
        k_stats<<<CC, 256, 0, s>>>(C_.conv, B * Ly, CC, C_.stat);
        k_bn_gelu<<<ceil_div(ty, 256), 256, 0, s>>>(C_.conv, C_.stat, C_.sb + i * CC,
                                                    C_.bb + i * CC, C_.zf, C_.zb, ty, CC);
        Lc = Ly;
    }
    int Lf = Lc;
    k_outconv<<<ceil_div(B * Lf, 256), 256, 0, s>>>(C_.zf, B * Lf, C_.w_out, C_.obuf);
    k_stats<<<1, 256, 0, s>>>(C_.obuf, B * Lf, 1, C_.stat);
    k_bn_final<<<ceil_div(B * Lf, 256), 256, 0, s>>>(C_.obuf, C_.stat, C_.s_out, C_.b_out,
                                                     dst, dstStride, dstOff, Lf);
}

extern "C" void kernel_launch(void* const* d_in, const int* in_sizes, int n_in,
                              void* d_out, int out_size, void* d_ws, size_t ws_size,
                              hipStream_t stream)
{
    (void)in_sizes; (void)n_in; (void)out_size; (void)ws_size;

    const float* x      = (const float*)d_in[0];
    Ctx C_;
    C_.w_in   = (const float*)d_in[1];
    const float* w0     = (const float*)d_in[2];
    const float* w_conv = (const float*)d_in[3];
    C_.w_skip = (const float*)d_in[4];
    C_.w_out  = (const float*)d_in[5];
    C_.s_in   = (const float*)d_in[6];
    C_.b_in   = (const float*)d_in[7];
    C_.s0a    = (const float*)d_in[8];
    C_.b0a    = (const float*)d_in[9];
    C_.s0b    = (const float*)d_in[10];
    C_.b0b    = (const float*)d_in[11];
    C_.sa     = (const float*)d_in[12];
    C_.ba     = (const float*)d_in[13];
    C_.sb     = (const float*)d_in[14];
    C_.bb     = (const float*)d_in[15];
    C_.s_out  = (const float*)d_in[16];
    C_.b_out  = (const float*)d_in[17];
    float* out = (float*)d_out;

    // workspace carve
    char* base = (char*)d_ws;
    size_t off = 0;
    auto carve = [&](size_t bytes) -> void* {
        void* p = base + off;
        off = (off + bytes + 255) & ~(size_t)255;
        return p;
    };
    C_.stat = (float*)carve(2 * CC * sizeof(float));
    unsigned short* wT0 = (unsigned short*)carve((size_t)CC * 14 * CC * 2);
    unsigned short* wTc = (unsigned short*)carve((size_t)15 * CC * 7 * CC * 2);
    C_.wT0 = wT0; C_.wTc = wTc;
    C_.conv = (float*)carve((size_t)B * 8192 * CC * 4);
    C_.zf   = (float*)carve((size_t)B * 8192 * CC * 4);
    C_.zb   = (unsigned short*)carve((size_t)B * 8192 * CC * 2);
    C_.obuf = (float*)carve((size_t)B * 4096 * 4);
    float* zval = (float*)carve(256);
    float* fbuf = (float*)carve((size_t)B * FSTRIDE * 4);
    float* win  = (float*)carve((size_t)B * ZLEN_ * 4);

    // weights -> bf16, transposed to [co][K]
    k_prep_w<<<ceil_div(14 * CC * CC, 256), 256, 0, stream>>>(w0, 14, wT0);
    for (int i = 0; i < 15; ++i)
        k_prep_w<<<ceil_div(7 * CC * CC, 256), 256, 0, stream>>>(
            w_conv + (size_t)i * 7 * CC * CC, 7, wTc + (size_t)i * CC * 7 * CC);

    // z0: full stack on L=8128 -> out[:, 0:3968]
    run_stack(stream, C_, x, 8192, 0, 8128, out, 4000, 0);

    // scan: 32 sequential single-sample steps on 194-sample windows
    k_init_fbuf<<<ceil_div(B * 8128, 256), 256, 0, stream>>>(x, out, fbuf);
    int cur_len = 8128;
    for (int step = 0; step < 32; ++step) {
        k_build_window<<<ceil_div(B * ZLEN_, 256), 256, 0, stream>>>(win, fbuf, cur_len, x, step);
        run_stack(stream, C_, win, ZLEN_, 0, ZLEN_, zval, 1, 0);
        k_commit<<<1, 32, 0, stream>>>(fbuf, cur_len, x, step, zval, out);
        cur_len += 2;
    }
}
